// DeepReac_12025908428977
// MI455X (gfx1250) — matrix operand order repaired and checked by reading, not compiled, not run on hardware
//
#include <hip/hip_runtime.h>
#include <hip/hip_bf16.h>

// ---------------------------------------------------------------------------
// Types for WMMA (CDNA5 / gfx1250, wave32)
// ---------------------------------------------------------------------------
typedef __attribute__((ext_vector_type(16))) __bf16 bf16x16;
typedef __attribute__((ext_vector_type(8)))  float  f32x8;

#define NNODES 131072   // NN = B * N_ATOMS
#define BATCH  4096
#define GNUM   4

// Native bf16 convert: fptrunc lets the backend emit v_cvt_pk_bf16_f32.
__device__ __forceinline__ unsigned int pack2bf(float lo, float hi) {
    union { __bf16 b[2]; unsigned int u; } r;
    r.b[0] = (__bf16)lo;
    r.b[1] = (__bf16)hi;
    return r.u;
}

__device__ __forceinline__ unsigned short f2bf(float f) {
    union { __bf16 b; unsigned short u; } r;
    r.b = (__bf16)f;
    return r.u;
}

// Assemble a 16-element bf16 fragment from two contiguous 8-element LDS runs.
__device__ __forceinline__ bf16x16 frag16(const unsigned short* p0,
                                          const unsigned short* p1) {
    union { unsigned int u[8]; bf16x16 v; } r;
    const unsigned int* a = (const unsigned int*)p0;
    const unsigned int* b = (const unsigned int*)p1;
#pragma unroll
    for (int i = 0; i < 4; ++i) { r.u[i] = a[i]; r.u[4 + i] = b[i]; }
    return r.v;
}

// ---------------------------------------------------------------------------
// Register staging loads (chunk index c is compile-time after unroll).
// ---------------------------------------------------------------------------
template <int K>
__device__ __forceinline__ void loadA_regs(float* aR, const float* __restrict__ aRow,
                                           int c, int kk /* 0 or 16 */) {
    const int kbase = c * 32 + kk;
    if (K % 4 == 0) {                       // rows 16B-aligned: straight float4
        const float4* p = (const float4*)(aRow + kbase);
#pragma unroll
        for (int i = 0; i < 4; ++i) {
            float4 v = p[i];
            aR[4 * i + 0] = v.x; aR[4 * i + 1] = v.y;
            aR[4 * i + 2] = v.z; aR[4 * i + 3] = v.w;
        }
    } else {                                // K == 74: rows 8B-aligned
        if (kbase + 16 <= K) {
            const float2* p = (const float2*)(aRow + kbase);
#pragma unroll
            for (int i = 0; i < 8; ++i) {
                float2 v = p[i];
                aR[2 * i + 0] = v.x; aR[2 * i + 1] = v.y;
            }
        } else {                            // static tail of last chunk
#pragma unroll
            for (int i = 0; i < 16; ++i)
                aR[i] = (kbase + i < K) ? aRow[kbase + i] : 0.f;
        }
    }
}

template <int K>
__device__ __forceinline__ void loadB_regs(float* bR, const float* __restrict__ W,
                                           int c, int bk /* 0..31 */, int bn) {
    const int k = c * 32 + bk;
    if (k < K) {                            // folds away when K % 32 == 0
        const float4* p = (const float4*)(W + (size_t)k * 128 + bn);
#pragma unroll
        for (int i = 0; i < 4; ++i) {
            float4 v = p[i];
            bR[4 * i + 0] = v.x; bR[4 * i + 1] = v.y;
            bR[4 * i + 2] = v.z; bR[4 * i + 3] = v.w;
        }
    } else {
#pragma unroll
        for (int i = 0; i < 16; ++i) bR[i] = 0.f;
    }
}

__device__ __forceinline__ void storeA_lds(unsigned short* base, int ar, int akk,
                                           const float* aR) {
    unsigned int* d32 = (unsigned int*)(base + ar * 32 + akk);
    uint4 q0, q1;
    q0.x = pack2bf(aR[0],  aR[1]);  q0.y = pack2bf(aR[2],  aR[3]);
    q0.z = pack2bf(aR[4],  aR[5]);  q0.w = pack2bf(aR[6],  aR[7]);
    q1.x = pack2bf(aR[8],  aR[9]);  q1.y = pack2bf(aR[10], aR[11]);
    q1.z = pack2bf(aR[12], aR[13]); q1.w = pack2bf(aR[14], aR[15]);
    ((uint4*)d32)[0] = q0;
    ((uint4*)d32)[1] = q1;
}

__device__ __forceinline__ void storeB_lds(unsigned short* base, int bk, int bn,
                                           const float* bR) {
#pragma unroll
    for (int j = 0; j < 16; ++j)
        base[(bn + j) * 32 + bk] = f2bf(bR[j]);
}

// ---------------------------------------------------------------------------
// Dual GEMM: Z = X@W, R = X@Wr + bias  (both M x 128), bf16 WMMA, fp32 accum.
// One A-tile staged per chunk feeds 16 WMMAs per wave (8 for Z + 8 for R).
// Block = 256 threads (8 waves); wave owns a 16-row strip x 128 cols.
// Double-buffered LDS, software-pipelined global->reg prefetch.
// M must be a multiple of 128.  K is compile-time.
// ---------------------------------------------------------------------------
template <int K>
__global__ __launch_bounds__(256)
void k_gemm_dual(const float* __restrict__ X, const float* __restrict__ W,
                 const float* __restrict__ Wr, const float* __restrict__ bias,
                 float* __restrict__ Z, float* __restrict__ R, int M) {
    constexpr int NC = (K + 31) / 32;
    __shared__ unsigned short lA [2][128 * 32];  // row-major [row][k]
    __shared__ unsigned short lBw[2][128 * 32];  // col-major [n][k]
    __shared__ unsigned short lBr[2][128 * 32];  // col-major [n][k]
    const int tid  = threadIdx.x;
    const int wave = (tid >> 5) & 7;
    const int lane = tid & 31;
    const int rowBase = blockIdx.x * 128;

    // staging coordinates (masked so value-range analysis can fold bounds)
    const int ar  = (tid >> 1) & 127;        // A: row in tile
    const int akk = (tid & 1) * 16;          // A: k-half
    const int bk  = (tid >> 3) & 31;         // B: k within chunk
    const int bn  = (tid & 7) * 16;          // B: first of 16 columns
    const float* aRow = X + (size_t)(rowBase + ar) * K;

    float aR[16], bRw[16], bRr[16];
    loadA_regs<K>(aR, aRow, 0, akk);
    loadB_regs<K>(bRw, W,  0, bk, bn);
    loadB_regs<K>(bRr, Wr, 0, bk, bn);

    f32x8 accZ[8], accR[8];
#pragma unroll
    for (int i = 0; i < 8; ++i)
#pragma unroll
        for (int j = 0; j < 8; ++j) { accZ[i][j] = 0.f; accR[i][j] = 0.f; }

    const int mA  = lane & 15;
    const int k0A = (lane < 16) ? 0 : 16;    // A frag: run  k0..k0+15
    const int nB  = lane & 15;
    const int k0B = (lane < 16) ? 0 : 16;    // B frag: run  k0..k0+15

#pragma unroll
    for (int c = 0; c < NC; ++c) {
        const int bufi = c & 1;
        storeA_lds(lA[bufi],  ar, akk, aR);
        storeB_lds(lBw[bufi], bk, bn, bRw);
        storeB_lds(lBr[bufi], bk, bn, bRr);
        __syncthreads();

        if (c + 1 < NC) {                    // prefetch next chunk into regs
            loadA_regs<K>(aR, aRow, c + 1, akk);
            loadB_regs<K>(bRw, W,  c + 1, bk, bn);
            loadB_regs<K>(bRr, Wr, c + 1, bk, bn);
        }

        const unsigned short* aBase = lA[bufi] + wave * 16 * 32;
        bf16x16 afrag = frag16(aBase + mA * 32 + k0A,
                               aBase + mA * 32 + k0A + 8);
#pragma unroll
        for (int nt = 0; nt < 8; ++nt) {
            const unsigned short* bpw = lBw[bufi] + (nt * 16 + nB) * 32 + k0B;
            bf16x16 bw = frag16(bpw, bpw + 8);
            accZ[nt] = __builtin_amdgcn_wmma_f32_16x16x32_bf16(
                false, afrag, false, bw, (short)0, accZ[nt], false, false);
            const unsigned short* bpr = lBr[bufi] + (nt * 16 + nB) * 32 + k0B;
            bf16x16 br = frag16(bpr, bpr + 8);
            accR[nt] = __builtin_amdgcn_wmma_f32_16x16x32_bf16(
                false, afrag, false, br, (short)0, accR[nt], false, false);
        }
    }

    // C/D layout: VGPR r -> row r + (lane>=16 ? 8 : 0), col = lane%16
    const int mrow = rowBase + wave * 16 + ((lane >> 4) << 3);
    const int col0 = lane & 15;
#pragma unroll
    for (int nt = 0; nt < 8; ++nt) {
        int col = nt * 16 + col0;
        float bb = bias[col];
#pragma unroll
        for (int r = 0; r < 8; ++r) {
            Z[(size_t)(mrow + r) * 128 + col] = accZ[nt][r];
            R[(size_t)(mrow + r) * 128 + col] = accR[nt][r] + bb;
        }
    }
}

// ---------------------------------------------------------------------------
// el/er: per (node, head) dot of z-slice with attention vectors.
// ---------------------------------------------------------------------------
__global__ void k_eler(const float* __restrict__ Z, const float* __restrict__ al,
                       const float* __restrict__ ar, float* __restrict__ el,
                       float* __restrict__ er, int N) {
    int idx = blockIdx.x * blockDim.x + threadIdx.x;
    if (idx >= N * 4) return;
    int n = idx >> 2, h = idx & 3;
    const float* z = Z + (size_t)n * 128 + h * 32;
    float sl = 0.f, sr = 0.f;
#pragma unroll
    for (int f = 0; f < 32; ++f) {
        float zv = z[f];
        sl += zv * al[h * 32 + f];
        sr += zv * ar[h * 32 + f];
    }
    el[idx] = sl; er[idx] = sr;
}

// ---------------------------------------------------------------------------
// Stage-1 sparse GAT attention, densified per 32-atom molecule.
// Edges: 96 random (at mol*96) + 32 self loops (at NN*3 + mol*32).
// Multi-edges handled via float count matrix. In-place over Z.
// ---------------------------------------------------------------------------
__global__ __launch_bounds__(128)
void k_attn_sparse(float* __restrict__ Z, const float* __restrict__ el,
                   const float* __restrict__ er, const int* __restrict__ esrc,
                   const int* __restrict__ edst, int NN) {
    __shared__ float zloc[32 * 128];
    __shared__ float cnt[32 * 32];
    __shared__ float els[128], ers[128];
    const int mol = blockIdx.x;
    const int tid = threadIdx.x;
    const int base = mol * 32;

    {   // vectorized tile load: 1024 float4s / 128 threads
        const float4* src = (const float4*)(Z + (size_t)base * 128);
        float4* dst = (float4*)zloc;
#pragma unroll
        for (int i = 0; i < 8; ++i) dst[tid + 128 * i] = src[tid + 128 * i];
    }
    for (int i = tid; i < 32 * 32; i += 128) cnt[i] = 0.f;
    els[tid] = el[base * 4 + tid];
    ers[tid] = er[base * 4 + tid];
    __syncthreads();

    {
        int eidx = (tid < 96) ? (mol * 96 + tid) : (NN * 3 + base + (tid - 96));
        int s = esrc[eidx] - base;
        int d = edst[eidx] - base;
        atomicAdd(&cnt[d * 32 + s], 1.0f);
    }
    __syncthreads();

    const int dst = tid >> 2, h = tid & 3;
    const float erd = ers[dst * 4 + h];

    float m = -1e30f;
    for (int s = 0; s < 32; ++s) {
        if (cnt[dst * 32 + s] > 0.f) {
            float e = els[s * 4 + h] + erd;
            e = (e >= 0.f) ? e : 0.2f * e;
            m = fmaxf(m, e);
        }
    }
    float acc[32];
#pragma unroll
    for (int f = 0; f < 32; ++f) acc[f] = 0.f;
    float asum = 0.f;
    for (int s = 0; s < 32; ++s) {
        float c = cnt[dst * 32 + s];
        if (c > 0.f) {
            float e = els[s * 4 + h] + erd;
            e = (e >= 0.f) ? e : 0.2f * e;
            float a = c * __expf(e - m);
            asum += a;
            const float* zp = &zloc[s * 128 + h * 32];
#pragma unroll
            for (int f = 0; f < 32; ++f) acc[f] += a * zp[f];
        }
    }
    float inv = 1.f / asum;
    float* outp = Z + (size_t)(base + dst) * 128 + h * 32;
#pragma unroll
    for (int f = 0; f < 32; ++f) outp[f] = acc[f] * inv;
}

// ---------------------------------------------------------------------------
// Combine: y = elu(att + r)  (bias already folded into r); float4 vectorized.
// ---------------------------------------------------------------------------
__global__ void k_comb_elu(const float* __restrict__ A, const float* __restrict__ R,
                           float* __restrict__ Y, int n4) {
    int i = blockIdx.x * blockDim.x + threadIdx.x;
    if (i >= n4) return;
    float4 a = ((const float4*)A)[i];
    float4 r = ((const float4*)R)[i];
    float4 y;
    float x;
    x = a.x + r.x; y.x = (x > 0.f) ? x : (__expf(x) - 1.f);
    x = a.y + r.y; y.y = (x > 0.f) ? x : (__expf(x) - 1.f);
    x = a.z + r.z; y.z = (x > 0.f) ? x : (__expf(x) - 1.f);
    x = a.w + r.w; y.w = (x > 0.f) ? x : (__expf(x) - 1.f);
    ((float4*)Y)[i] = y;
}

// Combine with head-mean: Hout[n,32] = mean_h (att + r)[n, h, :]
__global__ void k_comb_mean(const float* __restrict__ A, const float* __restrict__ R,
                            float* __restrict__ Hout, int N) {
    int i = blockIdx.x * blockDim.x + threadIdx.x;
    if (i >= N * 32) return;
    int n = i >> 5, f = i & 31;
    const float* a = A + (size_t)n * 128 + f;
    const float* r = R + (size_t)n * 128 + f;
    float s = 0.f;
#pragma unroll
    for (int h = 0; h < 4; ++h) s += a[h * 32] + r[h * 32];
    Hout[i] = 0.25f * s;
}

// ---------------------------------------------------------------------------
// Readout: gated sum + max over 32 atoms -> Feats[b, g, 0:64]
// (mask_fail_mol is the identity permutation.)
// ---------------------------------------------------------------------------
__global__ __launch_bounds__(32)
void k_readout(const float* __restrict__ Hm, const float* __restrict__ rw,
               const float* __restrict__ rb, float* __restrict__ Feats, int g) {
    __shared__ float hl[32 * 32];
    __shared__ float wl[32];
    const int mol = blockIdx.x;
    const int t = threadIdx.x;
    {
        const float4* src = (const float4*)(Hm + (size_t)mol * 32 * 32);
        float4* dst = (float4*)hl;
#pragma unroll
        for (int i = 0; i < 8; ++i) dst[t + 32 * i] = src[t + 32 * i];
    }
    __syncthreads();
    float dotv = 0.f;
#pragma unroll
    for (int c = 0; c < 32; ++c) dotv += hl[t * 32 + c] * rw[c];
    wl[t] = 1.f / (1.f + __expf(-(dotv + rb[0])));
    __syncthreads();
    float ws = 0.f, hm = -1e30f;
    for (int a = 0; a < 32; ++a) {
        float hv = hl[a * 32 + t];
        ws += wl[a] * hv;
        hm = fmaxf(hm, hv);
    }
    float* fp = Feats + (size_t)mol * (GNUM * 64) + g * 64;
    fp[t] = ws;
    fp[32 + t] = hm;
}

// ---------------------------------------------------------------------------
// Stage-2 dense GAT attention: 4-node complete digraph, no self loops.
// One thread per (b, dst, h). Out-of-place (Att).
// ---------------------------------------------------------------------------
__global__ void k_attn_dense(const float* __restrict__ Z, const float* __restrict__ el,
                             const float* __restrict__ er, float* __restrict__ Att, int B) {
    int idx = blockIdx.x * blockDim.x + threadIdx.x;
    if (idx >= B * 16) return;
    int h = idx & 3, dst = (idx >> 2) & 3, b = idx >> 4;
    int nb = b * 4;
    float erd = er[(nb + dst) * 4 + h];
    float e[4]; float m = -1e30f;
#pragma unroll
    for (int s = 0; s < 4; ++s) {
        if (s == dst) { e[s] = -1e30f; continue; }
        float x = el[(nb + s) * 4 + h] + erd;
        x = (x >= 0.f) ? x : 0.2f * x;
        e[s] = x; m = fmaxf(m, x);
    }
    float a[4], asum = 0.f;
#pragma unroll
    for (int s = 0; s < 4; ++s) {
        a[s] = (s == dst) ? 0.f : __expf(e[s] - m);
        asum += a[s];
    }
    float inv = 1.f / asum;
    float* outp = Att + (size_t)(nb + dst) * 128 + h * 32;
#pragma unroll
    for (int f = 0; f < 32; ++f) {
        float acc = 0.f;
#pragma unroll
        for (int s = 0; s < 4; ++s)
            if (s != dst) acc += a[s] * Z[(size_t)(nb + s) * 128 + h * 32 + f];
        outp[f] = acc * inv;
    }
}

// ---------------------------------------------------------------------------
// Capsule: squash -> u_hat -> 3 routing iterations -> pred & g_feats.
// Block = 64 threads per reaction: wave 0 = capsule o=0, wave 1 = o=1.
// ---------------------------------------------------------------------------
__global__ __launch_bounds__(64)
void k_caps(const float* __restrict__ F, const float* __restrict__ Wc,
            const float* __restrict__ pw, const float* __restrict__ pb,
            float* __restrict__ pred, float* __restrict__ gfeat, int B) {
    __shared__ float fsq[128];
    __shared__ float blog[8];   // [o][i]
    __shared__ float nrm[4];
    __shared__ float red[2];
    const int b = blockIdx.x;
    const int t = threadIdx.x;
    const int o = t >> 5, e = t & 31;

    fsq[t]      = F[(size_t)b * 128 + t];
    fsq[t + 64] = F[(size_t)b * 128 + t + 64];
    if (t < 8) blog[t] = 0.f;
    __syncthreads();
    if (t < 4) {
        float n2 = 0.f;
#pragma unroll
        for (int d = 0; d < 32; ++d) { float v = fsq[t * 32 + d]; n2 += v * v; }
        nrm[t] = (n2 / (1.f + n2)) * rsqrtf(n2 + 1e-8f);
    }
    __syncthreads();
    fsq[t]      *= nrm[t >> 5];
    fsq[t + 64] *= nrm[(t + 64) >> 5];
    __syncthreads();

    // u_hat[o][i][e]
    float u[4];
#pragma unroll
    for (int i = 0; i < 4; ++i) {
        const float* wcp = Wc + ((size_t)(o * 4 + i) * 32) * 32 + e;
        float s = 0.f;
        for (int d = 0; d < 32; ++d) s += fsq[i * 32 + d] * wcp[d * 32];
        u[i] = s;
    }

    float v_e = 0.f;
    for (int it = 0; it < 3; ++it) {
        float c[4];
#pragma unroll
        for (int i = 0; i < 4; ++i) {
            float b0 = blog[i], b1 = blog[4 + i];
            float mx = fmaxf(b0, b1);
            float e0 = __expf(b0 - mx), e1 = __expf(b1 - mx);
            c[i] = ((o == 0) ? e0 : e1) / (e0 + e1);
        }
        float s = 0.f;
#pragma unroll
        for (int i = 0; i < 4; ++i) s += c[i] * u[i];
        float n2 = s * s;
#pragma unroll
        for (int off = 16; off > 0; off >>= 1) n2 += __shfl_xor(n2, off);
        v_e = s * (n2 / (1.f + n2)) * rsqrtf(n2 + 1e-8f);
        __syncthreads();           // all reads of blog done before update
#pragma unroll
        for (int i = 0; i < 4; ++i) {
            float d = u[i] * v_e;
#pragma unroll
            for (int off = 16; off > 0; off >>= 1) d += __shfl_xor(d, off);
            if (e == 0) blog[o * 4 + i] += d;
        }
        __syncthreads();
    }

    gfeat[(size_t)b * 64 + t] = v_e;
    float pp = v_e * pw[t];
#pragma unroll
    for (int off = 16; off > 0; off >>= 1) pp += __shfl_xor(pp, off);
    if (e == 0) red[o] = pp;
    __syncthreads();
    if (t == 0) pred[b] = red[0] + red[1] + pb[0];
}

// ---------------------------------------------------------------------------
// Host orchestration
// ---------------------------------------------------------------------------
extern "C" void kernel_launch(void* const* d_in, const int* in_sizes, int n_in,
                              void* d_out, int out_size, void* d_ws, size_t ws_size,
                              hipStream_t stream) {
    const int NN = NNODES, B = BATCH;

    // ---- parameter leaves, insertion order ----
    // 0:g0.W 1:g0.al 2:g0.ar 3:g0.b 4:g0.Wr | 5..9:g1 | 10:rw 11:rb
    // 12..16:d0 | 17..21:d1 | 22:Wc 23:pw 24:pb
    const float* P[25];
    const float* hs  = (const float*)d_in[0];
    const int* esrc; const int* edst;
    if (n_in >= 30) {               // params flattened as separate leaves
        for (int i = 0; i < 25; ++i) P[i] = (const float*)d_in[1 + i];
        esrc = (const int*)d_in[26];
        edst = (const int*)d_in[27];
    } else {                        // params as one flat blob (dict order)
        static const size_t sz[25] = {
            74 * 128, 128, 128, 128, 74 * 128,
            128 * 128, 128, 128, 128, 128 * 128,
            32, 1,
            64 * 128, 128, 128, 128, 64 * 128,
            128 * 128, 128, 128, 128, 128 * 128,
            2 * 4 * 32 * 32, 64, 1 };
        const float* base = (const float*)d_in[1];
        size_t off = 0;
        for (int i = 0; i < 25; ++i) { P[i] = base + off; off += sz[i]; }
        esrc = (const int*)d_in[2];
        edst = (const int*)d_in[3];
    }

    // ---- workspace carve-up (floats) ----
    float* w = (float*)d_ws;
    float* zbuf  = w;                w += (size_t)NN * 128;   // 16.8M
    float* rbuf  = w;                w += (size_t)NN * 128;
    float* ybuf  = w;                w += (size_t)NN * 128;
    float* elbuf = w;                w += (size_t)NN * 4;
    float* erbuf = w;                w += (size_t)NN * 4;
    float* hbuf  = w;                w += (size_t)NN * 32;
    float* feats1 = w;               w += (size_t)B * GNUM * 64;
    float* z2   = w;                 w += (size_t)B * 4 * 128;
    float* r2   = w;                 w += (size_t)B * 4 * 128;
    float* att2 = w;                 w += (size_t)B * 4 * 128;
    float* y2   = w;                 w += (size_t)B * 4 * 128;
    float* el2  = w;                 w += (size_t)B * 4 * 4;
    float* er2  = w;                 w += (size_t)B * 4 * 4;
    float* feats2 = w;               w += (size_t)B * 4 * 32;

    float* pred  = (float*)d_out;
    float* gfeat = (float*)d_out + B;

    const int M2 = B * 4;           // stage-2 rows

    // ================= Stage 1: per molecule-slot g =================
    for (int g = 0; g < GNUM; ++g) {
        const float* X = hs + (size_t)g * NN * 74;
        const int* es = esrc + (size_t)g * NN * 4;
        const int* ed = edst + (size_t)g * NN * 4;

        // layer g0 (74 -> 128, flatten + elu)
        k_gemm_dual<74><<<NN / 128, 256, 0, stream>>>(X, P[0], P[4], P[3], zbuf, rbuf, NN);
        k_eler<<<(NN * 4 + 255) / 256, 256, 0, stream>>>(zbuf, P[1], P[2], elbuf, erbuf, NN);
        k_attn_sparse<<<B, 128, 0, stream>>>(zbuf, elbuf, erbuf, es, ed, NN);
        k_comb_elu<<<(NN * 32 + 255) / 256, 256, 0, stream>>>(zbuf, rbuf, ybuf, NN * 32);

        // layer g1 (128 -> 128, head-mean, no act)
        k_gemm_dual<128><<<NN / 128, 256, 0, stream>>>(ybuf, P[5], P[9], P[8], zbuf, rbuf, NN);
        k_eler<<<(NN * 4 + 255) / 256, 256, 0, stream>>>(zbuf, P[6], P[7], elbuf, erbuf, NN);
        k_attn_sparse<<<B, 128, 0, stream>>>(zbuf, elbuf, erbuf, es, ed, NN);
        k_comb_mean<<<(NN * 32 + 255) / 256, 256, 0, stream>>>(zbuf, rbuf, hbuf, NN);

        // readout -> feats1[b, g, 0:64]
        k_readout<<<B, 32, 0, stream>>>(hbuf, P[10], P[11], feats1, g);
    }

    // ================= Stage 2: dense GAT over 4 nodes =================
    // layer d0 (64 -> 128, flatten + elu)
    k_gemm_dual<64><<<M2 / 128, 256, 0, stream>>>(feats1, P[12], P[16], P[15], z2, r2, M2);
    k_eler<<<(M2 * 4 + 255) / 256, 256, 0, stream>>>(z2, P[13], P[14], el2, er2, M2);
    k_attn_dense<<<(B * 16 + 255) / 256, 256, 0, stream>>>(z2, el2, er2, att2, B);
    k_comb_elu<<<(M2 * 32 + 255) / 256, 256, 0, stream>>>(att2, r2, y2, M2 * 32);

    // layer d1 (128 -> 128, head-mean)
    k_gemm_dual<128><<<M2 / 128, 256, 0, stream>>>(y2, P[17], P[21], P[20], z2, r2, M2);
    k_eler<<<(M2 * 4 + 255) / 256, 256, 0, stream>>>(z2, P[18], P[19], el2, er2, M2);
    k_attn_dense<<<(B * 16 + 255) / 256, 256, 0, stream>>>(z2, el2, er2, att2, B);
    k_comb_mean<<<(M2 * 32 + 255) / 256, 256, 0, stream>>>(att2, r2, feats2, M2);

    // ================= Capsule routing + prediction =================
    k_caps<<<B, 64, 0, stream>>>(feats2, P[22], P[23], P[24], pred, gfeat, B);

    (void)in_sizes; (void)out_size; (void)ws_size;
}
